// LatentSDE_74371653697573
// MI455X (gfx1250) — matrix operand order repaired and checked by reading, not compiled, not run on hardware
//
#include <hip/hip_runtime.h>
#include <cmath>

#define T_STEPS 100
#define BATCH   2048
#define LOG2PI_F 1.8378770664093453f
#define CLAMP_F  6.0f
#define WAVES    4

typedef __attribute__((ext_vector_type(16))) _Float16 v16h;
typedef __attribute__((ext_vector_type(8)))  _Float16 v8h;
typedef __attribute__((ext_vector_type(8)))  float    v8f;

// ---------------- helpers ----------------
__device__ __forceinline__ float sp_f(float x) {
  // numerically stable softplus: max(x,0) + log1p(exp(-|x|))
  return fmaxf(x, 0.0f) + log1pf(__expf(-fabsf(x)));
}
__device__ __forceinline__ float clamp6(float x) {
  return fminf(fmaxf(x, -CLAMP_F), CLAMP_F);
}
__device__ __forceinline__ float sigm(float x) {
  return 1.0f / (1.0f + __expf(-x));
}
__device__ __forceinline__ v8f wmma16(v16h a, v16h b, v8f c) {
  // D = A(16x32 f16) * B(32x16 f16) + C(f32)
  return __builtin_amdgcn_wmma_f32_16x16x32_f16(false, a, false, b, (short)0, c,
                                                false, false);
}

// A fragment (16x32 f16) from LDS row-major [16][ldk]: two 16B vector loads.
__device__ __forceinline__ v16h frag_a(const _Float16* A, int lane, int kbase, int ldk) {
  int m   = lane & 15;
  int klo = (lane < 16) ? 0 : 8;
  const _Float16* row = A + m * ldk + kbase + klo;
  v8h lo = *(const v8h*)(row);
  v8h hi = *(const v8h*)(row + 16);
  return __builtin_shufflevector(lo, hi, 0, 1, 2, 3, 4, 5, 6, 7,
                                 8, 9, 10, 11, 12, 13, 14, 15);
}

// B fragment from pre-swizzled LDS: one 32B vector load per lane.
__device__ __forceinline__ v16h frag_b_sw(const _Float16* Wf, int lane, int fragIdx) {
  return *(const v16h*)(Wf + ((size_t)fragIdx * 32 + lane) * 16);
}

// Fill pre-swizzled B-fragment weights: Wf[fi][lane][j] = W[k][n] (f16), where
// fi = kc*NT + nt, k = kc*32 + (lane>=16 ? 8 : 0) + (j&7) + (j>=8 ? 16 : 0), n = nt*16 + (lane&15).
__device__ __forceinline__ void fill_wfrag(_Float16* Wf, const float* W, int K, int N,
                                           int NT, int nfrag, int tid, int nthreads) {
  for (int i = tid; i < nfrag * 512; i += nthreads) {
    int fi   = i >> 9;
    int lane = (i >> 4) & 31;
    int j    = i & 15;
    int kc   = fi / NT, nt = fi - kc * NT;
    int k = kc * 32 + ((lane >> 4) << 3) + (j & 7) + ((j >> 3) << 4);
    int n = nt * 16 + (lane & 15);
    Wf[i] = (_Float16)((k < K && n < N) ? W[k * N + n] : 0.0f);
  }
}

// ---------------- init ----------------
__global__ void ls_init_kernel(float* out) {
  if (threadIdx.x < 2) out[threadIdx.x] = 0.0f;
}

// ---------------- encoder: 150 -> 30 -> 30 -> 15 over T*B rows ----------------
// row r = t*B + b consumes xs[t,b,:,:] (150 contiguous f32).
// ctx[t,b,:] = out[12:15]; rows with t==0 give qz0_mean/logstd (out[0:6]/[6:12]).
__global__ __launch_bounds__(128) void ls_encoder_kernel(
    const float* __restrict__ xs,
    const float* __restrict__ eW1, const float* __restrict__ eb1,
    const float* __restrict__ eW2, const float* __restrict__ eb2,
    const float* __restrict__ eW3, const float* __restrict__ eb3,
    float* __restrict__ ctx, float* __restrict__ qm, float* __restrict__ ql) {
  __shared__ __align__(32) _Float16 W1f[10 * 512];  // 5 k-chunks x 2 n-tiles
  __shared__ __align__(32) _Float16 W2f[2 * 512];   // 1 x 2
  __shared__ __align__(32) _Float16 W3f[1 * 512];   // 1 x 1
  __shared__ float b1s[32], b2s[32], b3s[16];
  __shared__ __align__(32) _Float16 Ast[WAVES][16 * 160];
  __shared__ __align__(32) _Float16 Mid[WAVES][16 * 32];

  const int tid = threadIdx.x;
  fill_wfrag(W1f, eW1, 150, 30, 2, 10, tid, 128);
  fill_wfrag(W2f, eW2, 30, 30, 2, 2, tid, 128);
  fill_wfrag(W3f, eW3, 30, 15, 1, 1, tid, 128);
  if (tid < 32) {
    b1s[tid] = (tid < 30) ? eb1[tid] : 0.0f;
    b2s[tid] = (tid < 30) ? eb2[tid] : 0.0f;
  }
  if (tid < 16) b3s[tid] = (tid < 15) ? eb3[tid] : 0.0f;
  __syncthreads();

  const int wave = tid >> 5;
  const int lane = tid & 31;
  const int tile = blockIdx.x * WAVES + wave;
  const int row0 = tile * 16;
  _Float16* A = Ast[wave];
  _Float16* M = Mid[wave];

  // stage A tile: 16 rows x 160 (pad 150->160); one v8h LDS store per 8 elems
  for (int c = lane; c < 16 * 20; c += 32) {
    int m = c / 20, kb = (c - m * 20) * 8;
    const float* src = xs + (size_t)(row0 + m) * 150 + kb;
    v8h v;
#pragma unroll
    for (int j = 0; j < 8; ++j)
      v[j] = (_Float16)((kb + j < 150) ? src[j] : 0.0f);
    *(v8h*)(A + m * 160 + kb) = v;
  }
  __syncthreads();

  const int n  = lane & 15;
  const int mb = (lane < 16) ? 0 : 8;

  // layer 1: K=160 (5 chunks), N=32 (2 tiles)
  v8f acc0 = {}, acc1 = {};
#pragma unroll
  for (int kc = 0; kc < 5; ++kc) {
    v16h a  = frag_a(A, lane, kc * 32, 160);
    v16h bA = frag_b_sw(W1f, lane, kc * 2 + 0);
    v16h bB = frag_b_sw(W1f, lane, kc * 2 + 1);
    acc0 = wmma16(a, bA, acc0);
    acc1 = wmma16(a, bB, acc1);
  }
#pragma unroll
  for (int r = 0; r < 8; ++r) {
    M[(mb + r) * 32 + n]      = (_Float16)sp_f(acc0[r] + b1s[n]);
    M[(mb + r) * 32 + n + 16] = (_Float16)sp_f(acc1[r] + b1s[n + 16]);
  }
  __syncthreads();

  // layer 2: K=32, N=32
  {
    v16h a  = frag_a(M, lane, 0, 32);
    v16h bA = frag_b_sw(W2f, lane, 0);
    v16h bB = frag_b_sw(W2f, lane, 1);
    v8f c0 = {}, c1 = {};
    c0 = wmma16(a, bA, c0);
    c1 = wmma16(a, bB, c1);
    __syncthreads();
#pragma unroll
    for (int r = 0; r < 8; ++r) {
      M[(mb + r) * 32 + n]      = (_Float16)sp_f(c0[r] + b2s[n]);
      M[(mb + r) * 32 + n + 16] = (_Float16)sp_f(c1[r] + b2s[n + 16]);
    }
  }
  __syncthreads();

  // layer 3: K=32, N=16 (15 valid)
  {
    v16h a  = frag_a(M, lane, 0, 32);
    v16h bA = frag_b_sw(W3f, lane, 0);
    v8f c = {};
    c = wmma16(a, bA, c);
#pragma unroll
    for (int r = 0; r < 8; ++r) {
      float val = c[r] + b3s[n];
      int row = row0 + mb + r;
      int t = row >> 11;           // /BATCH
      int b = row & (BATCH - 1);
      if (n >= 12 && n < 15) ctx[((size_t)t * BATCH + b) * 3 + (n - 12)] = val;
      if (t == 0) {
        if (n < 6)       qm[b * 6 + n] = val;
        else if (n < 12) ql[b * 6 + (n - 6)] = val;
      }
    }
  }
}

// ---------------- SDE scan: sequential over 99 steps, one thread per sample --
__global__ __launch_bounds__(256) void ls_sde_kernel(
    const float* __restrict__ ts, const float* __restrict__ eps0,
    const float* __restrict__ dW,
    const float* __restrict__ fW1, const float* __restrict__ fb1,
    const float* __restrict__ fW2, const float* __restrict__ fb2,
    const float* __restrict__ hW1, const float* __restrict__ hb1,
    const float* __restrict__ hW2, const float* __restrict__ hb2,
    const float* __restrict__ gW1, const float* __restrict__ gb1,
    const float* __restrict__ gW2, const float* __restrict__ gb2,
    const float* __restrict__ pz0m, const float* __restrict__ pz0l,
    const float* __restrict__ qm, const float* __restrict__ ql,
    const float* __restrict__ ctx,
    float* __restrict__ zs, float* __restrict__ out) {
  __shared__ float sfW1[300], sfb1[30], sfW2[180], sfb2[6];
  __shared__ float shW1[210], shb1[30], shW2[180], shb2[6];
  __shared__ float sgW1[360], sgb1[180], sgW2[180], sgb2[6];
  __shared__ float sts[T_STEPS];

  const int tid = threadIdx.x, bd = blockDim.x;
  for (int i = tid; i < 300; i += bd) sfW1[i] = fW1[i];
  for (int i = tid; i < 30;  i += bd) sfb1[i] = fb1[i];
  for (int i = tid; i < 180; i += bd) sfW2[i] = fW2[i];
  for (int i = tid; i < 6;   i += bd) sfb2[i] = fb2[i];
  for (int i = tid; i < 210; i += bd) shW1[i] = hW1[i];
  for (int i = tid; i < 30;  i += bd) shb1[i] = hb1[i];
  for (int i = tid; i < 180; i += bd) shW2[i] = hW2[i];
  for (int i = tid; i < 6;   i += bd) shb2[i] = hb2[i];
  for (int i = tid; i < 360; i += bd) sgW1[i] = gW1[i];
  for (int i = tid; i < 180; i += bd) sgb1[i] = gb1[i];
  for (int i = tid; i < 180; i += bd) sgW2[i] = gW2[i];
  for (int i = tid; i < 6;   i += bd) sgb2[i] = gb2[i];
  for (int i = tid; i < T_STEPS; i += bd) sts[i] = ts[i];
  __syncthreads();

  const int b = blockIdx.x * blockDim.x + threadIdx.x;  // 0..2047

  float z[6];
  float kl = 0.0f;
#pragma unroll
  for (int d = 0; d < 6; ++d) {
    float mean = qm[b * 6 + d];
    float lst  = clamp6(ql[b * 6 + d]);
    float sq   = __expf(lst);
    z[d] = mean + sq * eps0[b * 6 + d];
    zs[(size_t)b * 6 + d] = z[d];
    float pls = clamp6(pz0l[d]);
    float spz = __expf(pls);
    float dm  = mean - pz0m[d];
    kl += (pls - lst) + (sq * sq + dm * dm) / (2.0f * spz * spz) - 0.5f;
  }
  atomicAdd(&out[1], kl * (1.0f / BATCH));

  float logr_sum = 0.0f;
  for (int k = 0; k < T_STEPS - 1; ++k) {
    float t   = sts[k];
    float dtk = sts[k + 1] - sts[k];
    const float* c = &ctx[((size_t)(k + 1) * BATCH + b) * 3];
    float in10[10] = {t, z[0], z[1], z[2], z[3], z[4], z[5], c[0], c[1], c[2]};
    float in7[7]   = {t, z[0], z[1], z[2], z[3], z[4], z[5]};

    float fo[6], ho[6];
#pragma unroll
    for (int o = 0; o < 6; ++o) { fo[o] = sfb2[o]; ho[o] = shb2[o]; }
    for (int j = 0; j < 30; ++j) {
      float hf = sfb1[j];
#pragma unroll
      for (int i = 0; i < 10; ++i) hf = fmaf(in10[i], sfW1[i * 30 + j], hf);
      hf = sp_f(hf);
      float hh = shb1[j];
#pragma unroll
      for (int i = 0; i < 7; ++i) hh = fmaf(in7[i], shW1[i * 30 + j], hh);
      hh = sp_f(hh);
#pragma unroll
      for (int o = 0; o < 6; ++o) {
        fo[o] = fmaf(hf, sfW2[j * 6 + o], fo[o]);
        ho[o] = fmaf(hh, shW2[j * 6 + o], ho[o]);
      }
    }

    float g[6];
#pragma unroll
    for (int d = 0; d < 6; ++d) {
      float acc = 0.0f;
      const float* w0 = &sgW1[(d * 2 + 0) * 30];
      const float* w1 = &sgW1[(d * 2 + 1) * 30];
      const float* bb = &sgb1[d * 30];
      const float* w2 = &sgW2[d * 30];
      for (int j = 0; j < 30; ++j) {
        float hj = sp_f(fmaf(t, w0[j], fmaf(z[d], w1[j], bb[j])));
        acc = fmaf(hj, w2[j], acc);
      }
      g[d] = sigm(acc + sgb2[d]);
    }

    float usum = 0.0f;
#pragma unroll
    for (int d = 0; d < 6; ++d) {
      float u = (fo[d] - ho[d]) / g[d];
      usum = fmaf(u, u, usum);
    }
    logr_sum += 0.5f * usum * dtk;

    const float* dwk = &dW[((size_t)k * BATCH + b) * 6];
#pragma unroll
    for (int d = 0; d < 6; ++d) {
      z[d] = z[d] + fo[d] * dtk + g[d] * dwk[d];
      zs[((size_t)(k + 1) * BATCH + b) * 6 + d] = z[d];
    }
  }
  atomicAdd(&out[1], logr_sum * (1.0f / BATCH));
}

// ---------------- decoder: 6 -> 30 -> 30 -> 100 + Gaussian loglik reduction --
__global__ __launch_bounds__(128) void ls_decoder_kernel(
    const float* __restrict__ xs, const float* __restrict__ zs,
    const float* __restrict__ dcW1, const float* __restrict__ dcb1,
    const float* __restrict__ dcW2, const float* __restrict__ dcb2,
    const float* __restrict__ dcW3, const float* __restrict__ dcb3,
    float* __restrict__ out) {
  __shared__ __align__(32) _Float16 W1f[2 * 512];   // 1 x 2
  __shared__ __align__(32) _Float16 W2f[2 * 512];   // 1 x 2
  __shared__ __align__(32) _Float16 W3f[7 * 512];   // 1 x 7
  __shared__ float b1s[32], b2s[32], b3s[112];
  __shared__ __align__(32) _Float16 Ast[WAVES][16 * 32];
  __shared__ __align__(32) _Float16 Mid[WAVES][16 * 32];
  __shared__ float Dls[WAVES][16 * 112];

  const int tid = threadIdx.x;
  fill_wfrag(W1f, dcW1, 6, 30, 2, 2, tid, 128);
  fill_wfrag(W2f, dcW2, 30, 30, 2, 2, tid, 128);
  fill_wfrag(W3f, dcW3, 30, 100, 7, 7, tid, 128);
  if (tid < 32) {
    b1s[tid] = (tid < 30) ? dcb1[tid] : 0.0f;
    b2s[tid] = (tid < 30) ? dcb2[tid] : 0.0f;
  }
  for (int i = tid; i < 112; i += 128) b3s[i] = (i < 100) ? dcb3[i] : 0.0f;
  __syncthreads();

  const int wave = tid >> 5;
  const int lane = tid & 31;
  const int tile = blockIdx.x * WAVES + wave;
  const int row0 = tile * 16;
  _Float16* A = Ast[wave];
  _Float16* M = Mid[wave];
  float*    D = Dls[wave];

  // stage A: 16 rows x 32 (6 valid features from zs), one v8h store per chunk
  for (int c = lane; c < 16 * 4; c += 32) {
    int m = c >> 2, kb = (c & 3) * 8;
    const float* src = zs + (size_t)(row0 + m) * 6;
    v8h v;
#pragma unroll
    for (int j = 0; j < 8; ++j)
      v[j] = (_Float16)((kb + j < 6) ? src[kb + j] : 0.0f);
    *(v8h*)(A + m * 32 + kb) = v;
  }
  __syncthreads();

  const int n  = lane & 15;
  const int mb = (lane < 16) ? 0 : 8;

  // layer 1
  {
    v16h a  = frag_a(A, lane, 0, 32);
    v16h bA = frag_b_sw(W1f, lane, 0);
    v16h bB = frag_b_sw(W1f, lane, 1);
    v8f c0 = {}, c1 = {};
    c0 = wmma16(a, bA, c0);
    c1 = wmma16(a, bB, c1);
#pragma unroll
    for (int r = 0; r < 8; ++r) {
      M[(mb + r) * 32 + n]      = (_Float16)sp_f(c0[r] + b1s[n]);
      M[(mb + r) * 32 + n + 16] = (_Float16)sp_f(c1[r] + b1s[n + 16]);
    }
  }
  __syncthreads();

  // layer 2
  {
    v16h a  = frag_a(M, lane, 0, 32);
    v16h bA = frag_b_sw(W2f, lane, 0);
    v16h bB = frag_b_sw(W2f, lane, 1);
    v8f c0 = {}, c1 = {};
    c0 = wmma16(a, bA, c0);
    c1 = wmma16(a, bB, c1);
    __syncthreads();
#pragma unroll
    for (int r = 0; r < 8; ++r) {
      M[(mb + r) * 32 + n]      = (_Float16)sp_f(c0[r] + b2s[n]);
      M[(mb + r) * 32 + n + 16] = (_Float16)sp_f(c1[r] + b2s[n + 16]);
    }
  }
  __syncthreads();

  // layer 3: N=112 (100 valid), 7 N-tiles
  {
    v16h a = frag_a(M, lane, 0, 32);
#pragma unroll
    for (int nt = 0; nt < 7; ++nt) {
      v16h bA = frag_b_sw(W3f, lane, nt);
      v8f c = {};
      c = wmma16(a, bA, c);
#pragma unroll
      for (int r = 0; r < 8; ++r)
        D[(mb + r) * 112 + nt * 16 + n] = c[r] + b3s[nt * 16 + n];
    }
  }
  __syncthreads();

  // Gaussian log-likelihood partial sum over the 16x50 tile
  float part = 0.0f;
  for (int i = lane; i < 16 * 50; i += 32) {
    int m = i / 50, j = i - m * 50;
    int row = row0 + m;
    float xm = D[m * 112 + j];
    float xl = clamp6(D[m * 112 + 50 + j]);
    float xsig = __expf(xl);
    float tgt = xs[((size_t)row * 3 + 2) * 50 + j];
    float dv = (tgt - xm) / xsig;
    part += -0.5f * dv * dv - xl - 0.5f * LOG2PI_F;
  }
#pragma unroll
  for (int off = 16; off > 0; off >>= 1) part += __shfl_xor(part, off, 32);
  if (lane == 0) atomicAdd(&out[0], part * (1.0f / BATCH));
}

// ---------------- launch ----------------
extern "C" void kernel_launch(void* const* d_in, const int* in_sizes, int n_in,
                              void* d_out, int out_size, void* d_ws, size_t ws_size,
                              hipStream_t stream) {
  const float* xs   = (const float*)d_in[0];
  const float* ts   = (const float*)d_in[1];
  const float* eps0 = (const float*)d_in[2];
  const float* dW   = (const float*)d_in[3];
  const float* eW1  = (const float*)d_in[4];
  const float* eb1  = (const float*)d_in[5];
  const float* eW2  = (const float*)d_in[6];
  const float* eb2  = (const float*)d_in[7];
  const float* eW3  = (const float*)d_in[8];
  const float* eb3  = (const float*)d_in[9];
  const float* dcW1 = (const float*)d_in[10];
  const float* dcb1 = (const float*)d_in[11];
  const float* dcW2 = (const float*)d_in[12];
  const float* dcb2 = (const float*)d_in[13];
  const float* dcW3 = (const float*)d_in[14];
  const float* dcb3 = (const float*)d_in[15];
  const float* fW1  = (const float*)d_in[16];
  const float* fb1  = (const float*)d_in[17];
  const float* fW2  = (const float*)d_in[18];
  const float* fb2  = (const float*)d_in[19];
  const float* hW1  = (const float*)d_in[20];
  const float* hb1  = (const float*)d_in[21];
  const float* hW2  = (const float*)d_in[22];
  const float* hb2  = (const float*)d_in[23];
  const float* gW1  = (const float*)d_in[24];
  const float* gb1  = (const float*)d_in[25];
  const float* gW2  = (const float*)d_in[26];
  const float* gb2  = (const float*)d_in[27];
  const float* pz0m = (const float*)d_in[28];
  const float* pz0l = (const float*)d_in[29];

  float* out = (float*)d_out;
  float* ws  = (float*)d_ws;
  float* ctx = ws;                                  // T*B*3 = 614400
  float* qm  = ctx + (size_t)T_STEPS * BATCH * 3;   // B*6   = 12288
  float* ql  = qm + (size_t)BATCH * 6;
  float* zs  = ql + (size_t)BATCH * 6;              // T*B*6 = 1228800

  ls_init_kernel<<<1, 32, 0, stream>>>(out);

  // T*B / 16 rows-per-tile / 4 waves-per-block = 3200 blocks
  ls_encoder_kernel<<<3200, 128, 0, stream>>>(xs, eW1, eb1, eW2, eb2, eW3, eb3,
                                              ctx, qm, ql);

  ls_sde_kernel<<<BATCH / 256, 256, 0, stream>>>(
      ts, eps0, dW, fW1, fb1, fW2, fb2, hW1, hb1, hW2, hb2,
      gW1, gb1, gW2, gb2, pz0m, pz0l, qm, ql, ctx, zs, out);

  ls_decoder_kernel<<<3200, 128, 0, stream>>>(xs, zs, dcW1, dcb1, dcW2, dcb2,
                                              dcW3, dcb3, out);
}
